// BaseDirectedNet_51539608033
// MI455X (gfx1250) — matrix operand order repaired and checked by reading, not compiled, not run on hardware
//
#include <hip/hip_runtime.h>
#include <hip/hip_bf16.h>

typedef _Float16 half_t;
typedef __attribute__((ext_vector_type(16))) _Float16 v16h;
typedef __attribute__((ext_vector_type(8)))  _Float16 v8h;
typedef __attribute__((ext_vector_type(8)))  float    v8f;

union V16 { v16h v; v8h h[2]; };

#define NB     8192
#define NNODE  30
#define FB     5
#define INC    256
#define NFLT   256
#define NCLS   9
#define XSTR   40   // halves; 80B row stride -> 16B aligned, 16-bank spread

// ---- one-time weight transform: W[k][n] f32 -> Wt[n][k] f16 (B-operand friendly)
__global__ __launch_bounds__(256) void convert_weights_kernel(
    const float* __restrict__ W1, const float* __restrict__ W2,
    half_t* __restrict__ W1t, half_t* __restrict__ W2t) {
  const int n = blockIdx.x;   // 0..255
  const int k = threadIdx.x;  // 0..255
  W1t[n * 256 + k] = (half_t)W1[k * 256 + n];
  W2t[n * 256 + k] = (half_t)W2[k * 256 + n];
}

__global__ __launch_bounds__(256) void fused_gcn_kernel(
    const float* __restrict__ real, const float* __restrict__ graph,
    const half_t* __restrict__ W1t, const float* __restrict__ b1,
    const half_t* __restrict__ W2t, const float* __restrict__ b2,
    const float* __restrict__ wlin, const float* __restrict__ blin,
    const float* __restrict__ Whead, const float* __restrict__ bhead,
    float* __restrict__ out) {
  __shared__ __align__(16) half_t adjS[32 * 32];     // 2 KB, zero-padded 30->32
  __shared__ __align__(16) half_t bufA[256 * XSTR];  // 20 KB: xT -> h1T -> h2 rows
  __shared__ __align__(16) half_t bufB[32 * 256];    // 16 KB: t -> t2 (row-major)
  __shared__ float xlS[32];

  const int tid  = threadIdx.x;
  const int lane = tid & 31;
  const int wave = tid >> 5;
  const int r    = lane & 15;   // row (A/C) or col (B/C) within tile
  const int hi   = lane >> 4;   // half-wave select
  const size_t b = (size_t)blockIdx.x;

  // ---------- stage adj (mean over bands) and xT ----------
  for (int p = tid; p < 32 * 32; p += 256) adjS[p] = (half_t)0.f;
  __syncthreads();

  const float* gsrc = graph + b * (size_t)(FB * NNODE * NNODE);
  for (int p = tid; p < NNODE * NNODE; p += 256) {
    float s = 0.f;
#pragma unroll
    for (int f = 0; f < FB; ++f) s += gsrc[f * (NNODE * NNODE) + p];
    adjS[(p / NNODE) * 32 + (p % NNODE)] = (half_t)(s * 0.2f);
  }
  {  // xT: bufA[f*XSTR + j] = x[j][f], pad j=30,31
    const float* xsrc = real + b * (size_t)(NNODE * INC);
    half_t* row = &bufA[tid * XSTR];
#pragma unroll 5
    for (int j = 0; j < NNODE; ++j) row[j] = (half_t)xsrc[j * 256 + tid];
    row[30] = (half_t)0.f;
    row[31] = (half_t)0.f;
  }
  __syncthreads();

  // A operand: 16x32 f16, lane holds row (base%16); K = hi*16 .. hi*16+15 contiguous
  auto loadA = [&](const half_t* base, int row, int stride, int kofs) -> v16h {
    V16 a;
    const half_t* p = base + row * stride + kofs;
    a.h[0] = *(const v8h*)(p + hi * 16);
    a.h[1] = *(const v8h*)(p + hi * 16 + 8);
    return a.v;
  };
  // B operand: 32x16 f16, lane holds column; K = hi*16 .. hi*16+15 contiguous
  auto loadB = [&](const half_t* base, int col, int stride, int kofs) -> v16h {
    V16 bb;
    const half_t* p = base + col * stride + kofs + hi * 16;
    bb.h[0] = *(const v8h*)(p);
    bb.h[1] = *(const v8h*)(p + 8);
    return bb.v;
  };

  // ---------- GEMM1: t = adj @ x  (K=32, one WMMA per tile) ----------
  v8f accT[4];
#pragma unroll
  for (int t4 = 0; t4 < 4; ++t4) {
    const int tile = wave * 4 + t4, mt = tile >> 4, nt = tile & 15;
    v16h a  = loadA(adjS, mt * 16 + r, 32, 0);
    v16h bb = loadB(bufA, nt * 16 + r, XSTR, 0);
    v8f  c  = {};
    accT[t4] = __builtin_amdgcn_wmma_f32_16x16x32_f16(false, a, false, bb,
                                                      (short)0, c, false, false);
  }
#pragma unroll
  for (int t4 = 0; t4 < 4; ++t4) {
    const int tile = wave * 4 + t4, mt = tile >> 4, nt = tile & 15;
    const int m0 = mt * 16 + hi * 8, n = nt * 16 + r;
#pragma unroll
    for (int v = 0; v < 8; ++v) bufB[(m0 + v) * 256 + n] = (half_t)accT[t4][v];
  }
  __syncthreads();

  // ---------- GEMM2: h1 = relu(t @ W1 + b1) -> h1T in bufA ----------
#pragma unroll
  for (int t4 = 0; t4 < 4; ++t4) {
    const int tile = wave * 4 + t4, mt = tile >> 4, nt = tile & 15;
    v8f acc = {};
#pragma unroll
    for (int kt = 0; kt < 8; ++kt) {
      v16h a = loadA(bufB, mt * 16 + r, 256, kt * 32);
      const half_t* wp = W1t + (nt * 16 + r) * 256 + kt * 32 + hi * 16;
      V16 bb;
      bb.h[0] = *(const v8h*)(wp);
      bb.h[1] = *(const v8h*)(wp + 8);
      acc = __builtin_amdgcn_wmma_f32_16x16x32_f16(false, a, false, bb.v,
                                                   (short)0, acc, false, false);
    }
    const int f = nt * 16 + r, m0 = mt * 16 + hi * 8;
    const float bias = b1[f];
#pragma unroll
    for (int v = 0; v < 8; ++v)
      bufA[f * XSTR + m0 + v] = (half_t)fmaxf(acc[v] + bias, 0.f);
  }
  __syncthreads();

  // ---------- GEMM3: t2 = adj @ h1 (reads h1T in bufA) ----------
  v8f accU[4];
#pragma unroll
  for (int t4 = 0; t4 < 4; ++t4) {
    const int tile = wave * 4 + t4, mt = tile >> 4, nt = tile & 15;
    v16h a  = loadA(adjS, mt * 16 + r, 32, 0);
    v16h bb = loadB(bufA, nt * 16 + r, XSTR, 0);
    v8f  c  = {};
    accU[t4] = __builtin_amdgcn_wmma_f32_16x16x32_f16(false, a, false, bb,
                                                      (short)0, c, false, false);
  }
#pragma unroll
  for (int t4 = 0; t4 < 4; ++t4) {
    const int tile = wave * 4 + t4, mt = tile >> 4, nt = tile & 15;
    const int m0 = mt * 16 + hi * 8, n = nt * 16 + r;
#pragma unroll
    for (int v = 0; v < 8; ++v) bufB[(m0 + v) * 256 + n] = (half_t)accU[t4][v];
  }
  __syncthreads();

  // ---------- GEMM4: h2 = relu(t2 @ W2 + b2) -> row-major in bufA ----------
#pragma unroll
  for (int t4 = 0; t4 < 4; ++t4) {
    const int tile = wave * 4 + t4, mt = tile >> 4, nt = tile & 15;
    v8f acc = {};
#pragma unroll
    for (int kt = 0; kt < 8; ++kt) {
      v16h a = loadA(bufB, mt * 16 + r, 256, kt * 32);
      const half_t* wp = W2t + (nt * 16 + r) * 256 + kt * 32 + hi * 16;
      V16 bb;
      bb.h[0] = *(const v8h*)(wp);
      bb.h[1] = *(const v8h*)(wp + 8);
      acc = __builtin_amdgcn_wmma_f32_16x16x32_f16(false, a, false, bb.v,
                                                   (short)0, acc, false, false);
    }
    const int f = nt * 16 + r, m0 = mt * 16 + hi * 8;
    const float bias = b2[f];
#pragma unroll
    for (int v = 0; v < 8; ++v)
      bufA[(m0 + v) * 256 + f] = (half_t)fmaxf(acc[v] + bias, 0.f);
  }
  __syncthreads();

  // ---------- xl = relu(h2 @ w_lin + b_lin); out = xl @ W_head^T + b_head ----------
  if (tid < 32) xlS[tid] = blin[0];
  __syncthreads();
  {
    const int i = tid & 31, c = tid >> 5;  // 8 partials per node row
    const half_t* hrow = &bufA[i * 256 + c * 32];
    const float* wl = wlin + c * 32;
    float s = 0.f;
#pragma unroll 8
    for (int u = 0; u < 32; ++u) s += (float)hrow[u] * wl[u];
    atomicAdd(&xlS[i], s);
  }
  __syncthreads();
  if (tid < NCLS) {
    float o = bhead[tid];
    const float* wh = Whead + tid * NNODE;
#pragma unroll 6
    for (int i2 = 0; i2 < NNODE; ++i2) o += fmaxf(xlS[i2], 0.f) * wh[i2];
    out[b * NCLS + tid] = o;
  }
}

extern "C" void kernel_launch(void* const* d_in, const int* in_sizes, int n_in,
                              void* d_out, int out_size, void* d_ws, size_t ws_size,
                              hipStream_t stream) {
  (void)in_sizes; (void)n_in; (void)out_size; (void)ws_size;
  const float* real  = (const float*)d_in[0];
  /* d_in[1] = imag: unused by reference forward */
  const float* graph = (const float*)d_in[2];
  const float* W1    = (const float*)d_in[3];
  const float* b1    = (const float*)d_in[4];
  const float* W2    = (const float*)d_in[5];
  const float* b2    = (const float*)d_in[6];
  const float* wlin  = (const float*)d_in[7];
  const float* blin  = (const float*)d_in[8];
  const float* Whead = (const float*)d_in[9];
  const float* bhead = (const float*)d_in[10];

  half_t* W1t = (half_t*)d_ws;            // 256*256 f16 = 128 KB
  half_t* W2t = W1t + 256 * 256;          // next 128 KB

  convert_weights_kernel<<<256, 256, 0, stream>>>(W1, W2, W1t, W2t);
  fused_gcn_kernel<<<NB, 256, 0, stream>>>(real, graph, W1t, b1, W2t, b2,
                                           wlin, blin, Whead, bhead,
                                           (float*)d_out);
}
